// RNN_20418274525773
// MI455X (gfx1250) — matrix operand order repaired and checked by reading, not compiled, not run on hardware
//
#include <hip/hip_runtime.h>
#include <cstdint>

#define BB 32
#define TT 2048
#define II 12
#define HH 512
#define G3 1536
#define OO 3
#define NWG 16          // persistent workgroups in the recurrent scan
#define JW 32           // hidden columns owned per workgroup
#define LDS_STRIDE 520  // padded bf16 element stride (1024B row + 16B pad)

typedef __attribute__((ext_vector_type(16))) __bf16 v16bf;
typedef __attribute__((ext_vector_type(8)))  __bf16 v8bf;
typedef __attribute__((ext_vector_type(8)))  float  v8f;
typedef uint32_t u32x4 __attribute__((ext_vector_type(4)));
typedef int      i32x4 __attribute__((ext_vector_type(4)));
typedef int      i32x8 __attribute__((ext_vector_type(8)));

// Pointer pointee types for the async-load builtin (exact match per diagnostic:
// gcc-vector int4, addrspace(1) source / addrspace(3) LDS destination).
typedef int v4i_raw __attribute__((vector_size(16)));
typedef __attribute__((address_space(1))) v4i_raw glob_v4i;
typedef __attribute__((address_space(3))) v4i_raw lds_v4i;

// Tensor Data Mover builtin: 5-arg form on clang-22 (ROCm 7.2); 6-arg on clang-23.
#if defined(__clang_major__) && \
    __has_builtin(__builtin_amdgcn_tensor_load_to_lds) && \
    __has_builtin(__builtin_amdgcn_s_wait_tensorcnt)
#define TDM_OK 1
#if (__clang_major__ >= 23)
#define TDM_6ARG 1
#else
#define TDM_6ARG 0
#endif
#else
#define TDM_OK 0
#endif

#if __has_builtin(__builtin_amdgcn_global_load_async_to_lds_b128)
#define ASYNC_OK 1
#else
#define ASYNC_OK 0
#endif

__device__ inline __bf16 f2bf(float f) {
    uint32_t u = __builtin_bit_cast(uint32_t, f);
    uint32_t r = u + 0x7FFFu + ((u >> 16) & 1u);
    uint16_t h = (uint16_t)(r >> 16);
    return __builtin_bit_cast(__bf16, h);
}
__device__ inline float bf2f(__bf16 b) {
    uint16_t h = __builtin_bit_cast(uint16_t, b);
    uint32_t u = ((uint32_t)h) << 16;
    return __builtin_bit_cast(float, u);
}
__device__ inline v8f vzero8f() {
    v8f v = {0.f, 0.f, 0.f, 0.f, 0.f, 0.f, 0.f, 0.f};
    return v;
}

// ---- async global->LDS 16B move (ASYNCcnt-tracked), sync fallback ----
__device__ inline void async_cp16(void* lds, const void* g) {
#if ASYNC_OK
    __builtin_amdgcn_global_load_async_to_lds_b128(
        (glob_v4i*)(uintptr_t)g, (lds_v4i*)(uintptr_t)lds, 0, 0);
#else
    *(v8bf*)lds = *(const v8bf*)g;
#endif
}
__device__ inline void wait_async_le6() {
#if ASYNC_OK
#if __has_builtin(__builtin_amdgcn_s_wait_asynccnt)
    __builtin_amdgcn_s_wait_asynccnt(6);
#else
    asm volatile("s_wait_asynccnt 0x6" ::: "memory");
#endif
#endif
}
__device__ inline void wait_async_0() {
#if ASYNC_OK
#if __has_builtin(__builtin_amdgcn_s_wait_asynccnt)
    __builtin_amdgcn_s_wait_asynccnt(0);
#else
    asm volatile("s_wait_asynccnt 0x0" ::: "memory");
#endif
#endif
}

#if TDM_OK
// TDM DMA: copy 32x1024B tile (h state, bf16 32x512) global->LDS with 16B pad per
// row so the LDS layout lands at the padded LDS_STRIDE (520 elems) directly.
__device__ inline void tdm_load_h(const __bf16* hsrc, __bf16* h_lds) {
    uint64_t ga = (uint64_t)(uintptr_t)hsrc;
    uint32_t la = (uint32_t)(uintptr_t)h_lds;
    u32x4 g0;
    g0[0] = 1u;                                               // count=1 (valid user D#)
    g0[1] = la;                                               // lds_addr (bytes)
    g0[2] = (uint32_t)ga;                                     // global_addr[31:0]
    g0[3] = (uint32_t)((ga >> 32) & 0x01FFFFFFu) | (2u << 30);// addr[56:32] | type=2
    i32x8 g1;
    g1[0] = (3 << 16)    // data_size = 8B
          | (1 << 20)    // pad_enable
          | (7 << 22)    // pad_interval = 256 DWORDs (1024B rows)
          | (3 << 25);   // pad_amount   = 4 DWORDs (16B)
    g1[1] = (int)(128u << 16);   // tensor_dim0 = 128 (8B units) -> bits[79:48]
    g1[2] = (int)(32u << 16);    // tensor_dim1 = 32            -> bits[111:80]
    g1[3] = (int)(128u << 16);   // tile_dim0   = 128           -> bits[127:112]
    g1[4] = 32;                  // tile_dim1   = 32
    g1[5] = 128;                 // tensor_dim0_stride = 128 (8B units)
    g1[6] = 0;
    g1[7] = 0;
    i32x4 z4 = {0, 0, 0, 0};
#if TDM_6ARG
    i32x8 z8 = {0, 0, 0, 0, 0, 0, 0, 0};
    __builtin_amdgcn_tensor_load_to_lds(g0, g1, z4, z4, z8, 0);
#else
    __builtin_amdgcn_tensor_load_to_lds(g0, g1, z4, z4, 0);
#endif
    __builtin_amdgcn_s_wait_tensorcnt(0);
}
#endif

// D = A*B + C, bf16 in, fp32 accumulate (v_wmma_f32_16x16x32_bf16)
__device__ inline v8f wmma_bf16(v16bf a, v16bf b, v8f c) {
    return __builtin_amdgcn_wmma_f32_16x16x32_bf16(false, a, false, b, (short)0, c, false, false);
}

// A fragment (16x32 MxK): lane<16 -> K[0..7],K[16..23]; lane>=16 -> K[8..15],K[24..31]
__device__ inline v16bf ld_fragA(const __bf16* p, int rowBase, int stride, int k0) {
    int lane = threadIdx.x & 31;
    int half = lane >> 4;
    int row  = rowBase + (lane & 15);
    const __bf16* q = p + row * stride + k0 + half * 8;
    v8bf lo = *(const v8bf*)q;
    v8bf hi = *(const v8bf*)(q + 16);
    return __builtin_shufflevector(lo, hi, 0,1,2,3,4,5,6,7,8,9,10,11,12,13,14,15);
}
// B fragment (32x16 KxN): lane<16 -> K[0..7],K[16..23] of col n; lane>=16 -> K[8..15],K[24..31]
__device__ inline v16bf ld_fragB(const __bf16* p, int colBase, int stride, int k0) {
    int lane  = threadIdx.x & 31;
    int khalf = lane >> 4;
    int n     = colBase + (lane & 15);
    const __bf16* q = p + n * stride + k0 + khalf * 8;
    v8bf lo = *(const v8bf*)q;
    v8bf hi = *(const v8bf*)(q + 16);
    return __builtin_shufflevector(lo, hi, 0,1,2,3,4,5,6,7,8,9,10,11,12,13,14,15);
}

// ---------------- small utility kernels ----------------

__global__ void cvt_bf16_kernel(const float* __restrict__ in, __bf16* __restrict__ out, int n) {
    int i = blockIdx.x * blockDim.x + threadIdx.x;
    if (i < n) out[i] = f2bf(in[i]);
}

__global__ void xw0_kernel(const float* __restrict__ x, const float* __restrict__ w,
                           const float* __restrict__ bias, float* __restrict__ xw) {
    long long idx = (long long)blockIdx.x * blockDim.x + threadIdx.x;
    if (idx >= (long long)BB * TT * G3) return;
    int col = (int)(idx % G3);
    long long row = idx / G3;
    const float* xr = x + row * II;
    const float* wr = w + (long long)col * II;
    float s = bias[col];
#pragma unroll
    for (int i = 0; i < II; ++i) s += xr[i] * wr[i];
    xw[idx] = s;
}

__global__ void h0_kernel(const float* __restrict__ vel_init, const float* __restrict__ vdw,
                          const float* __restrict__ vdb, __bf16* __restrict__ h0) {
    int idx = blockIdx.x * blockDim.x + threadIdx.x;
    if (idx >= BB * HH) return;
    int b = idx >> 9, j = idx & (HH - 1);
    const float* v = vel_init + b * 3;
    const float* w = vdw + j * 3;
    h0[idx] = f2bf(v[0] * w[0] + v[1] * w[1] + v[2] * w[2] + vdb[j]);
}

// ---------------- big input-projection GEMM (layers 1,2) ----------------
// C[65536][1536] = A[65536][512](bf16) @ W[1536][512]^T(bf16) + bias (fp32 out)
// BK=64, double-buffered LDS, async staging overlapped with 8 WMMAs/stage/wave.
__global__ __launch_bounds__(256) void gemm_xw_kernel(const __bf16* __restrict__ A,
                                                      const __bf16* __restrict__ Wt,
                                                      const float* __restrict__ bias,
                                                      float* __restrict__ Cout) {
    __shared__ __bf16 As[2][64 * 72];
    __shared__ __bf16 Bs[2][128 * 72];
    int bm = blockIdx.x & 1023;
    int bn = blockIdx.x >> 10;
    int m0 = bm * 64, n0 = bn * 128;
    int tid = threadIdx.x;
    int wave = tid >> 5, lane = tid & 31;
    int mw = wave >> 2, nw = wave & 3;
    int half = lane >> 4;

    auto stage = [&](int ks, int buf) {
        int k0 = ks * 64;
#pragma unroll
        for (int kc = 0; kc < 2; ++kc) {          // A: 64x64 elems, 512 chunks
            int c = tid + 256 * kc;
            int r = c >> 3, c8 = (c & 7) * 8;
            async_cp16(&As[buf][r * 72 + c8], &A[(size_t)(m0 + r) * HH + k0 + c8]);
        }
#pragma unroll
        for (int kc = 0; kc < 4; ++kc) {          // B: 128x64 elems, 1024 chunks
            int c = tid + 256 * kc;
            int r = c >> 3, c8 = (c & 7) * 8;
            async_cp16(&Bs[buf][r * 72 + c8], &Wt[(size_t)(n0 + r) * HH + k0 + c8]);
        }
    };

    v8f acc00 = vzero8f(), acc01 = vzero8f(), acc10 = vzero8f(), acc11 = vzero8f();
    stage(0, 0);
    for (int ks = 0; ks < 8; ++ks) {
        int buf = ks & 1;
        if (ks + 1 < 8) {
            stage(ks + 1, buf ^ 1);   // prefetch next stage into other buffer
            wait_async_le6();         // in-order: current stage's 6 ops have landed
        } else {
            wait_async_0();
        }
        __syncthreads();
#pragma unroll
        for (int kk = 0; kk < 64; kk += 32) {
            v16bf a0 = ld_fragA(&As[buf][0], mw * 32 + 0,  72, kk);
            v16bf a1 = ld_fragA(&As[buf][0], mw * 32 + 16, 72, kk);
            v16bf b0 = ld_fragB(&Bs[buf][0], nw * 32 + 0,  72, kk);
            v16bf b1 = ld_fragB(&Bs[buf][0], nw * 32 + 16, 72, kk);
            acc00 = wmma_bf16(a0, b0, acc00);
            acc01 = wmma_bf16(a0, b1, acc01);
            acc10 = wmma_bf16(a1, b0, acc10);
            acc11 = wmma_bf16(a1, b1, acc11);
        }
        __syncthreads();              // protect buffer about to be re-staged
    }

    v8f* accs[4] = {&acc00, &acc01, &acc10, &acc11};
#pragma unroll
    for (int mi = 0; mi < 2; ++mi)
#pragma unroll
        for (int ni = 0; ni < 2; ++ni) {
            v8f a = *accs[mi * 2 + ni];
            int col = n0 + nw * 32 + ni * 16 + (lane & 15);
            float bv = bias[col];
#pragma unroll
            for (int i = 0; i < 8; ++i) {
                int row = m0 + mw * 32 + mi * 16 + half * 8 + i;
                Cout[(size_t)row * G3 + col] = a[i] + bv;
            }
        }
}

// ---------------- persistent recurrent scan ----------------
__global__ __launch_bounds__(128) void gru_scan_kernel(const float* __restrict__ xw,
                                                       const float* __restrict__ b_hh,
                                                       const __bf16* __restrict__ whh,
                                                       __bf16* __restrict__ hbuf,   // [2][B*H]
                                                       __bf16* __restrict__ ybf,    // [B*T][H]
                                                       float* __restrict__ yf,      // optional fp32 copy
                                                       int* __restrict__ cnt,
                                                       int write_f32) {
    extern __shared__ char smem_raw[];
    __bf16* h_lds = (__bf16*)smem_raw;               // [32][LDS_STRIDE]
    __bf16* w_lds = h_lds + 32 * LDS_STRIDE;         // [96][LDS_STRIDE]

    const int wg = blockIdx.x;
    const int tid = threadIdx.x;
    const int wave = tid >> 5, lane = tid & 31;
    const int m = wave >> 1, s = wave & 1;           // batch half, j-subtile
    const int half = lane >> 4;
    const int j = wg * JW + s * 16 + (lane & 15);    // this lane's hidden column

    // load W_hh slice (96 x 512 bf16 = 96KB) once, async
    for (int c = tid; c < 96 * 64; c += 128) {
        int r = c >> 6;
        int c8 = (c & 63) * 8;
        int g = r >> 5, jl = r & 31;
        async_cp16(&w_lds[r * LDS_STRIDE + c8],
                   &whh[((size_t)g * HH + wg * JW + jl) * HH + c8]);
    }
    const float bR = b_hh[j], bZ = b_hh[HH + j], bN = b_hh[2 * HH + j];

    // fp32 private h state for the columns this wave produces
    v8f hstate = vzero8f();
#pragma unroll
    for (int i = 0; i < 8; ++i) {
        int b = m * 16 + half * 8 + i;
        hstate[i] = bf2f(hbuf[b * HH + j]);
    }
    wait_async_0();
    __syncthreads();

    for (int t = 0; t < TT; ++t) {
        const __bf16* hsrc = hbuf + (size_t)(t & 1) * (BB * HH);
        __bf16* hdst = hbuf + (size_t)((t + 1) & 1) * (BB * HH);

        // broadcast h (32x512 bf16 = 32KB) into LDS
#if TDM_OK
        if (tid < 32) tdm_load_h(hsrc, h_lds);       // wave 0 drives the TDM DMA
#else
        for (int c = tid; c < 32 * 64; c += 128) {
            int r = c >> 6, c8 = (c & 63) * 8;
            async_cp16(&h_lds[r * LDS_STRIDE + c8], &hsrc[r * HH + c8]);
        }
        wait_async_0();
#endif
        __syncthreads();

        // prefetch next step's gate pre-activations (global_prefetch_b8)
        if (t + 1 < TT) {
            int b0 = m * 16 + half * 8;
            const float* nx = xw + ((size_t)b0 * TT + t + 1) * G3;
            __builtin_prefetch(nx + j, 0, 1);
            __builtin_prefetch(nx + HH + j, 0, 1);
            __builtin_prefetch(nx + 2 * HH + j, 0, 1);
        }

        v8f accR = vzero8f(), accZ = vzero8f(), accN = vzero8f();
#pragma unroll 4
        for (int k0 = 0; k0 < HH; k0 += 32) {
            v16bf a  = ld_fragA(h_lds, m * 16, LDS_STRIDE, k0);
            v16bf br = ld_fragB(w_lds, 0 * JW + s * 16, LDS_STRIDE, k0);
            v16bf bz = ld_fragB(w_lds, 1 * JW + s * 16, LDS_STRIDE, k0);
            v16bf bn = ld_fragB(w_lds, 2 * JW + s * 16, LDS_STRIDE, k0);
            accR = wmma_bf16(a, br, accR);
            accZ = wmma_bf16(a, bz, accZ);
            accN = wmma_bf16(a, bn, accN);
        }

        // fused GRU elementwise + state update
#pragma unroll
        for (int i = 0; i < 8; ++i) {
            int b = m * 16 + half * 8 + i;
            const float* xr = xw + ((size_t)b * TT + t) * G3;
            float r = 1.f / (1.f + __expf(-(xr[j] + accR[i] + bR)));
            float z = 1.f / (1.f + __expf(-(xr[HH + j] + accZ[i] + bZ)));
            float n = tanhf(xr[2 * HH + j] + r * (accN[i] + bN));
            float hnew = (1.f - z) * n + z * hstate[i];
            hstate[i] = hnew;
            __bf16 hb = f2bf(hnew);
            hdst[b * HH + j] = hb;
            size_t yi = ((size_t)b * TT + t) * HH + j;
            ybf[yi] = hb;
            if (write_f32) yf[yi] = hnew;
        }

        // one grid barrier per step (double-buffered h makes this sufficient)
        __syncthreads();
        if (tid == 0) {
            __builtin_amdgcn_s_cluster_barrier();   // NOP when not launched as a cluster
            __threadfence();
            __hip_atomic_fetch_add(cnt, 1, __ATOMIC_RELEASE, __HIP_MEMORY_SCOPE_AGENT);
            int target = (t + 1) * NWG;
            while (__hip_atomic_load(cnt, __ATOMIC_ACQUIRE, __HIP_MEMORY_SCOPE_AGENT) < target)
                __builtin_amdgcn_s_sleep(1);
        }
        __syncthreads();
    }
}

// ---------------- epilogue ----------------

__global__ __launch_bounds__(256) void score_kernel(const float* __restrict__ y2,
                                                    const float* __restrict__ attn_w,
                                                    const float* __restrict__ attn_b,
                                                    const float* __restrict__ a1,
                                                    float* __restrict__ scores) {
    int wv = (int)((blockIdx.x * blockDim.x + threadIdx.x) >> 5);  // = b*T + t
    if (wv >= BB * TT) return;
    int lane = threadIdx.x & 31;
    const float* yr = y2 + (size_t)wv * HH;
    float slope = a1[0];
    float sum = 0.f;
    for (int k = lane; k < HH; k += 32) {
        float v = yr[k];
        v = v >= 0.f ? v : slope * v;
        sum += v * attn_w[k];
    }
#pragma unroll
    for (int off = 16; off; off >>= 1) sum += __shfl_xor(sum, off, 32);
    if (lane == 0) scores[wv] = sum + attn_b[0];
}

__global__ __launch_bounds__(256) void softmax_kernel(const float* __restrict__ scores,
                                                      float* __restrict__ attn) {
    __shared__ float red[256];
    int b = blockIdx.x;
    const float* s = scores + (size_t)b * TT;
    float mx = -1e30f;
    for (int t = threadIdx.x; t < TT; t += 256) mx = fmaxf(mx, s[t]);
    red[threadIdx.x] = mx; __syncthreads();
    for (int o = 128; o; o >>= 1) {
        if ((int)threadIdx.x < o) red[threadIdx.x] = fmaxf(red[threadIdx.x], red[threadIdx.x + o]);
        __syncthreads();
    }
    mx = red[0]; __syncthreads();
    float sm = 0.f;
    for (int t = threadIdx.x; t < TT; t += 256) sm += __expf(s[t] - mx);
    red[threadIdx.x] = sm; __syncthreads();
    for (int o = 128; o; o >>= 1) {
        if ((int)threadIdx.x < o) red[threadIdx.x] += red[threadIdx.x + o];
        __syncthreads();
    }
    float inv = 1.f / red[0];
    for (int t = threadIdx.x; t < TT; t += 256) attn[(size_t)b * TT + t] = __expf(s[t] - mx) * inv;
}

__global__ __launch_bounds__(128) void final_kernel(const float* __restrict__ y2,
                                                    const float* __restrict__ x,
                                                    const float* __restrict__ attn,
                                                    const float* __restrict__ a1,
                                                    const float* __restrict__ dec_w,
                                                    const float* __restrict__ dec_b,
                                                    const float* __restrict__ vel_w,
                                                    const float* __restrict__ vel_b,
                                                    const float* __restrict__ std_w,
                                                    const float* __restrict__ std_b,
                                                    float* __restrict__ out) {
    __shared__ float xs[II];
    __shared__ float red[4][6];
    int row = blockIdx.x;       // b*T + t
    int tid = threadIdx.x;
    if (tid < II) xs[tid] = x[(size_t)row * II + tid];
    __syncthreads();
    float at = attn[row];
    float slope = a1[0];
    float p[6] = {0.f, 0.f, 0.f, 0.f, 0.f, 0.f};
    const float* yr = y2 + (size_t)row * HH;
    for (int jj = tid; jj < HH; jj += 128) {
        float v = yr[jj];
        v = v >= 0.f ? v : slope * v;
        const float* dw = dec_w + (size_t)jj * II;
        float d = dec_b[jj];
#pragma unroll
        for (int i = 0; i < II; ++i) d += xs[i] * dw[i];
        float o = v + at * d;
#pragma unroll
        for (int k = 0; k < 3; ++k) {
            p[k]     += o * vel_w[k * HH + jj];
            p[3 + k] += o * std_w[k * HH + jj];
        }
    }
#pragma unroll
    for (int k = 0; k < 6; ++k) {
        float v = p[k];
#pragma unroll
        for (int off = 16; off; off >>= 1) v += __shfl_xor(v, off, 32);
        p[k] = v;
    }
    int wv = tid >> 5, lane = tid & 31;
    if (lane == 0) {
#pragma unroll
        for (int k = 0; k < 6; ++k) red[wv][k] = p[k];
    }
    __syncthreads();
    if (tid < 6) {
        float v = red[0][tid] + red[1][tid] + red[2][tid] + red[3][tid];
        int k = tid % 3;
        if (tid < 3) out[(size_t)row * 3 + k] = v + vel_b[k];
        else         out[(size_t)BB * TT * 3 + (size_t)row * 3 + k] = v + std_b[k];
    }
}

// ---------------- host orchestration ----------------

extern "C" void kernel_launch(void* const* d_in, const int* in_sizes, int n_in,
                              void* d_out, int out_size, void* d_ws, size_t ws_size,
                              hipStream_t stream) {
    (void)in_sizes; (void)n_in; (void)out_size; (void)ws_size;
    const float* x        = (const float*)d_in[0];
    const float* vel_init = (const float*)d_in[1];
    const float* w_ih[3]  = {(const float*)d_in[2], (const float*)d_in[6],  (const float*)d_in[10]};
    const float* w_hh[3]  = {(const float*)d_in[3], (const float*)d_in[7],  (const float*)d_in[11]};
    const float* b_ih[3]  = {(const float*)d_in[4], (const float*)d_in[8],  (const float*)d_in[12]};
    const float* b_hh[3]  = {(const float*)d_in[5], (const float*)d_in[9],  (const float*)d_in[13]};
    const float* a1       = (const float*)d_in[14];
    const float* dec_w    = (const float*)d_in[15];
    const float* dec_b    = (const float*)d_in[16];
    const float* vd_w[3]  = {(const float*)d_in[17], (const float*)d_in[19], (const float*)d_in[21]};
    const float* vd_b[3]  = {(const float*)d_in[18], (const float*)d_in[20], (const float*)d_in[22]};
    const float* vel_w    = (const float*)d_in[23];
    const float* vel_b    = (const float*)d_in[24];
    const float* std_w    = (const float*)d_in[25];
    const float* std_b    = (const float*)d_in[26];
    const float* attn_w   = (const float*)d_in[27];
    const float* attn_b   = (const float*)d_in[28];
    float* outp = (float*)d_out;

    char* ws = (char*)d_ws;
    size_t off = 0;
    auto alloc = [&](size_t bytes) -> void* {
        void* p = ws + off;
        off = (off + bytes + 255) & ~(size_t)255;
        return p;
    };
    float*  XW    = (float*) alloc((size_t)BB * TT * G3 * 4);   // 402 MB, reused per layer
    __bf16* YA    = (__bf16*)alloc((size_t)BB * TT * HH * 2);
    __bf16* YB    = (__bf16*)alloc((size_t)BB * TT * HH * 2);
    float*  Y2F   = (float*) alloc((size_t)BB * TT * HH * 4);
    __bf16* WIHB1 = (__bf16*)alloc((size_t)G3 * HH * 2);
    __bf16* WIHB2 = (__bf16*)alloc((size_t)G3 * HH * 2);
    __bf16* WHHB  = (__bf16*)alloc((size_t)3 * G3 * HH * 2);
    __bf16* HBUF  = (__bf16*)alloc((size_t)2 * BB * HH * 2);
    float*  SC    = (float*) alloc((size_t)BB * TT * 4);
    float*  AT    = (float*) alloc((size_t)BB * TT * 4);
    int*    CNT   = (int*)   alloc(256);

    const int nW = G3 * HH;
    cvt_bf16_kernel<<<(nW + 255) / 256, 256, 0, stream>>>(w_ih[1], WIHB1, nW);
    cvt_bf16_kernel<<<(nW + 255) / 256, 256, 0, stream>>>(w_ih[2], WIHB2, nW);
    for (int l = 0; l < 3; ++l)
        cvt_bf16_kernel<<<(nW + 255) / 256, 256, 0, stream>>>(w_hh[l], WHHB + (size_t)l * nW, nW);

    {
        long long n = (long long)BB * TT * G3;
        xw0_kernel<<<(unsigned)((n + 255) / 256), 256, 0, stream>>>(x, w_ih[0], b_ih[0], XW);
    }

    const size_t scan_lds = (size_t)(32 + 96) * LDS_STRIDE * sizeof(__bf16);  // ~130 KB of 320 KB
    (void)hipFuncSetAttribute(reinterpret_cast<const void*>(gru_scan_kernel),
                              hipFuncAttributeMaxDynamicSharedMemorySize, (int)scan_lds);

    // layer 0
    (void)hipMemsetAsync(CNT, 0, 256, stream);
    h0_kernel<<<(BB * HH + 255) / 256, 256, 0, stream>>>(vel_init, vd_w[0], vd_b[0], HBUF);
    gru_scan_kernel<<<NWG, 128, scan_lds, stream>>>(XW, b_hh[0], WHHB, HBUF, YA, nullptr, CNT, 0);

    // layer 1
    gemm_xw_kernel<<<1024 * 12, 256, 0, stream>>>(YA, WIHB1, b_ih[1], XW);
    (void)hipMemsetAsync(CNT, 0, 256, stream);
    h0_kernel<<<(BB * HH + 255) / 256, 256, 0, stream>>>(vel_init, vd_w[1], vd_b[1], HBUF);
    gru_scan_kernel<<<NWG, 128, scan_lds, stream>>>(XW, b_hh[1], WHHB + (size_t)nW, HBUF, YB, nullptr, CNT, 0);

    // layer 2
    gemm_xw_kernel<<<1024 * 12, 256, 0, stream>>>(YB, WIHB2, b_ih[2], XW);
    (void)hipMemsetAsync(CNT, 0, 256, stream);
    h0_kernel<<<(BB * HH + 255) / 256, 256, 0, stream>>>(vel_init, vd_w[2], vd_b[2], HBUF);
    gru_scan_kernel<<<NWG, 128, scan_lds, stream>>>(XW, b_hh[2], WHHB + (size_t)2 * nW, HBUF, YA, Y2F, CNT, 1);

    // epilogue
    score_kernel<<<(BB * TT) / 8, 256, 0, stream>>>(Y2F, attn_w, attn_b, a1, SC);
    softmax_kernel<<<BB, 256, 0, stream>>>(SC, AT);
    final_kernel<<<BB * TT, 128, 0, stream>>>(Y2F, x, AT, a1, dec_w, dec_b,
                                              vel_w, vel_b, std_w, std_b, outp);
}
